// SimpleMoE_88682484727935
// MI455X (gfx1250) — compile-verified, matrix operand order checked by reading, not run on hardware
//
#include <hip/hip_runtime.h>

// ---------------- problem constants ----------------
#define VOCAB   32000
#define DMODEL  1024
#define HIDDEN  2048
#define NEXP    8
#define NTOK    2048          // B*S = 2*1024
#define LN_EPS  1e-5f

typedef __bf16 bf16;
typedef bf16  v16bf  __attribute__((ext_vector_type(16)));
typedef bf16  bf16x8 __attribute__((ext_vector_type(8)));
typedef bf16  bf16x4 __attribute__((ext_vector_type(4)));
typedef float v8f    __attribute__((ext_vector_type(8)));

// Load a 16-element bf16 WMMA fragment (A or B matrix) for one lane.
// Per CDNA5 ISA 7.12.2 (16-bit A 16x32): vector elems 0..7 hold K = base..base+7,
// elems 8..15 hold K = base+16..base+23, where base = kb + (lane>>4)*8 and the
// row (M for A, N for B) = lane&15.  Caller passes rowptr already offset to
// row*ld + kb + g*8.  Two 16-byte loads.
__device__ __forceinline__ v16bf load_frag(const bf16* __restrict__ p) {
  bf16x8 lo = *(const bf16x8*)(p);
  bf16x8 hi = *(const bf16x8*)(p + 16);
  v16bf f;
#pragma unroll
  for (int i = 0; i < 8; ++i) { f[i] = lo[i]; f[i + 8] = hi[i]; }
  return f;
}

__device__ __forceinline__ v8f wmma_bf16(v16bf a, v16bf b, v8f c) {
  return __builtin_amdgcn_wmma_f32_16x16x32_bf16(false, a, false, b, (short)0, c,
                                                 false, false);
}

// ---------------- kernel 1: f32 -> bf16 transpose-convert ----------------
// in:  [R, C] f32 row-major (per batch)   out: [C, R] bf16 row-major (K contig)
__global__ __launch_bounds__(256) void moe_transpose_cvt(
    const float* __restrict__ in, bf16* __restrict__ out, int R, int C) {
  __shared__ float tile[16][17];
  const size_t bofs = (size_t)blockIdx.z * (size_t)R * (size_t)C;
  in  += bofs;
  out += bofs;
  const int c0 = blockIdx.x * 16, r0 = blockIdx.y * 16;
  const int tx = threadIdx.x, ty = threadIdx.y;
  tile[ty][tx] = in[(size_t)(r0 + ty) * C + (c0 + tx)];
  __syncthreads();
  out[(size_t)(c0 + ty) * R + (r0 + tx)] = (bf16)tile[tx][ty];
}

// ---------------- kernel 2: embed gather + gate softmax/top-2 ----------------
__global__ __launch_bounds__(256) void moe_embed_gate(
    const int* __restrict__ x, const float* __restrict__ embed,
    const float* __restrict__ Wg, const float* __restrict__ bg,
    bf16* __restrict__ hbf, float* __restrict__ gatew) {
  __shared__ float red[256][8];
  __shared__ float gl[8];
  const int t   = blockIdx.x;
  const int tid = threadIdx.x;
  const int tok = x[t];
  const float* row = embed + (size_t)tok * DMODEL;
  const int d = tid * 4;
  const float4 v = *(const float4*)(row + d);
  bf16x4 hb; hb[0] = (bf16)v.x; hb[1] = (bf16)v.y; hb[2] = (bf16)v.z; hb[3] = (bf16)v.w;
  *(bf16x4*)(hbf + (size_t)t * DMODEL + d) = hb;

  float p[8];
#pragma unroll
  for (int e = 0; e < 8; ++e) p[e] = 0.f;
  const float vv[4] = {v.x, v.y, v.z, v.w};
#pragma unroll
  for (int j = 0; j < 4; ++j) {
    const float* wr = Wg + (size_t)(d + j) * NEXP;
#pragma unroll
    for (int e = 0; e < 8; ++e) p[e] += vv[j] * wr[e];
  }
#pragma unroll
  for (int e = 0; e < 8; ++e) red[tid][e] = p[e];
  __syncthreads();
  if (tid < 8) {
    float s = 0.f;
    for (int i = 0; i < 256; ++i) s += red[i][tid];
    gl[tid] = s + bg[tid];
  }
  __syncthreads();
  if (tid == 0) {
    float m = gl[0];
#pragma unroll
    for (int e = 1; e < 8; ++e) m = fmaxf(m, gl[e]);
    float ex[8], sum = 0.f;
#pragma unroll
    for (int e = 0; e < 8; ++e) { ex[e] = expf(gl[e] - m); sum += ex[e]; }
    const float inv = 1.f / sum;
    int i1 = 0; float v1 = ex[0] * inv;
#pragma unroll
    for (int e = 1; e < 8; ++e) { float pe = ex[e] * inv; if (pe > v1) { v1 = pe; i1 = e; } }
    int i2 = -1; float v2 = -1.f;
#pragma unroll
    for (int e = 0; e < 8; ++e) {
      if (e == i1) continue;
      float pe = ex[e] * inv;
      if (pe > v2) { v2 = pe; i2 = e; }
    }
    float w[8];
#pragma unroll
    for (int e = 0; e < 8; ++e) w[e] = 0.f;
    w[i1] = v1; w[i2] = v2;
#pragma unroll
    for (int e = 0; e < 8; ++e) gatew[(size_t)t * NEXP + e] = w[e];
  }
}

// ---------------- kernel 3: dense masked MoE FFN (WMMA bf16) ----------------
// grid (NTOK/16, NEXP), 256 threads (8 waves).
// Phase A: act[16,2048] = relu(h_tile @ W1[e] + b1) -> LDS (bf16).
// Phase B: y[16,1024] = act @ W2[e] + b2, scaled by gate weight, atomicAdd.
__global__ __launch_bounds__(256) void moe_ffn(
    const bf16* __restrict__ hbf, const bf16* __restrict__ w1t,
    const bf16* __restrict__ w2t, const float* __restrict__ b1,
    const float* __restrict__ b2, const float* __restrict__ gatew,
    float* __restrict__ combined) {
  __shared__ __align__(16) bf16 sAct[16 * HIDDEN];  // 64 KB
  const int t0   = blockIdx.x * 16;
  const int e    = blockIdx.y;
  const int wv   = threadIdx.x >> 5;   // wave32
  const int lane = threadIdx.x & 31;
  const int g    = lane >> 4;
  const int l15  = lane & 15;

  // ---- Phase A : 16 hidden tiles per wave (256 cols) ----
  const bf16* aBase  = hbf + (size_t)(t0 + l15) * DMODEL + g * 8;
  const bf16* w1Base = w1t + (size_t)e * HIDDEN * DMODEL;
  v8f acc[16];
#pragma unroll
  for (int nt = 0; nt < 16; ++nt) {
    const float bv = b1[e * HIDDEN + wv * 256 + nt * 16 + l15];
#pragma unroll
    for (int r = 0; r < 8; ++r) acc[nt][r] = bv;
  }
#pragma unroll 1
  for (int kb = 0; kb < DMODEL; kb += 32) {
    const v16bf a = load_frag(aBase + kb);
#pragma unroll
    for (int nt = 0; nt < 16; ++nt) {
      const int n = wv * 256 + nt * 16 + l15;
      const v16bf b = load_frag(w1Base + (size_t)n * DMODEL + kb + g * 8);
      acc[nt] = wmma_bf16(a, b, acc[nt]);
    }
  }
#pragma unroll
  for (int nt = 0; nt < 16; ++nt) {
    const int n = wv * 256 + nt * 16 + l15;
#pragma unroll
    for (int r = 0; r < 8; ++r) {
      float vr = acc[nt][r];
      vr = vr > 0.f ? vr : 0.f;                 // relu
      sAct[(r + 8 * g) * HIDDEN + n] = (bf16)vr;
    }
  }
  __syncthreads();

  // ---- Phase B : 8 d_model tiles per wave (128 cols) ----
  const bf16* w2Base = w2t + (size_t)e * DMODEL * HIDDEN;
  const bf16* aL     = sAct + (size_t)l15 * HIDDEN + g * 8;
  v8f acc2[8];
#pragma unroll
  for (int nt = 0; nt < 8; ++nt) {
    const float bv = b2[e * DMODEL + wv * 128 + nt * 16 + l15];
#pragma unroll
    for (int r = 0; r < 8; ++r) acc2[nt][r] = bv;
  }
#pragma unroll 1
  for (int kb = 0; kb < HIDDEN; kb += 32) {
    const v16bf a = load_frag(aL + kb);         // LDS reads
#pragma unroll
    for (int nt = 0; nt < 8; ++nt) {
      const int n = wv * 128 + nt * 16 + l15;
      const v16bf b = load_frag(w2Base + (size_t)n * HIDDEN + kb + g * 8);
      acc2[nt] = wmma_bf16(a, b, acc2[nt]);
    }
  }
  float wrow[8];
#pragma unroll
  for (int r = 0; r < 8; ++r)
    wrow[r] = gatew[(size_t)(t0 + r + 8 * g) * NEXP + e];
#pragma unroll
  for (int nt = 0; nt < 8; ++nt) {
    const int n = wv * 128 + nt * 16 + l15;
#pragma unroll
    for (int r = 0; r < 8; ++r) {
      const float wgt = wrow[r];
      if (wgt != 0.f)
        atomicAdd(&combined[(size_t)(t0 + r + 8 * g) * DMODEL + n],
                  wgt * acc2[nt][r]);
    }
  }
}

// ---------------- kernel 4: LayerNorm -> bf16 ----------------
__global__ __launch_bounds__(256) void moe_layernorm(
    const float* __restrict__ combined, const float* __restrict__ gamma,
    const float* __restrict__ beta, bf16* __restrict__ lnbf) {
  __shared__ float s1[256], s2[256];
  const int t = blockIdx.x, tid = threadIdx.x;
  const float* rowp = combined + (size_t)t * DMODEL;
  const int d = tid * 4;
  const float4 v = *(const float4*)(rowp + d);
  s1[tid] = v.x + v.y + v.z + v.w;
  s2[tid] = v.x * v.x + v.y * v.y + v.z * v.z + v.w * v.w;
  __syncthreads();
  for (int off = 128; off > 0; off >>= 1) {
    if (tid < off) { s1[tid] += s1[tid + off]; s2[tid] += s2[tid + off]; }
    __syncthreads();
  }
  const float mu   = s1[0] * (1.f / DMODEL);
  const float var  = s2[0] * (1.f / DMODEL) - mu * mu;
  const float rstd = rsqrtf(var + LN_EPS);
  bf16x4 o;
  o[0] = (bf16)((v.x - mu) * rstd * gamma[d + 0] + beta[d + 0]);
  o[1] = (bf16)((v.y - mu) * rstd * gamma[d + 1] + beta[d + 1]);
  o[2] = (bf16)((v.z - mu) * rstd * gamma[d + 2] + beta[d + 2]);
  o[3] = (bf16)((v.w - mu) * rstd * gamma[d + 3] + beta[d + 3]);
  *(bf16x4*)(lnbf + (size_t)t * DMODEL + d) = o;
}

// ---------------- kernel 5: LM head GEMM (WMMA bf16) ----------------
// grid (NTOK/128, VOCAB/256), 256 threads = 8 waves in a 2x4 grid.
// Each wave computes a 64x64 region as 4x4 16x16 WMMA tiles:
// per K-step 8 fragment loads feed 16 WMMAs (0.5 fragment/WMMA).
__global__ __launch_bounds__(256) void moe_head(
    const bf16* __restrict__ lnbf, const bf16* __restrict__ whT,
    const float* __restrict__ bh, float* __restrict__ out) {
  const int wv   = threadIdx.x >> 5;
  const int lane = threadIdx.x & 31;
  const int g    = lane >> 4;
  const int l15  = lane & 15;
  const int mrow = wv >> 2;             // 0..1
  const int ncol = wv & 3;              // 0..3
  const int t0   = blockIdx.x * 128 + mrow * 64;
  const int c0   = blockIdx.y * 256 + ncol * 64;

  v8f acc[4][4];
#pragma unroll
  for (int mi = 0; mi < 4; ++mi)
#pragma unroll
    for (int ni = 0; ni < 4; ++ni)
#pragma unroll
      for (int r = 0; r < 8; ++r) acc[mi][ni][r] = 0.f;

  const bf16* aBase = lnbf + (size_t)(t0 + l15) * DMODEL + g * 8;
  const bf16* bBase = whT + (size_t)(c0 + l15) * DMODEL + g * 8;
#pragma unroll 1
  for (int kb = 0; kb < DMODEL; kb += 32) {
    v16bf A[4];
#pragma unroll
    for (int mi = 0; mi < 4; ++mi)
      A[mi] = load_frag(aBase + (size_t)(mi * 16) * DMODEL + kb);
#pragma unroll
    for (int ni = 0; ni < 4; ++ni) {
      const v16bf B = load_frag(bBase + (size_t)(ni * 16) * DMODEL + kb);
#pragma unroll
      for (int mi = 0; mi < 4; ++mi)
        acc[mi][ni] = wmma_bf16(A[mi], B, acc[mi][ni]);
    }
  }
#pragma unroll
  for (int ni = 0; ni < 4; ++ni) {
    const int col = c0 + ni * 16 + l15;
    const float bias = bh[col];
#pragma unroll
    for (int mi = 0; mi < 4; ++mi) {
#pragma unroll
      for (int r = 0; r < 8; ++r) {
        const int row = t0 + mi * 16 + r + 8 * g;
        out[(size_t)row * VOCAB + col] = acc[mi][ni][r] + bias;
      }
    }
  }
}

// ---------------- host-side launcher ----------------
extern "C" void kernel_launch(void* const* d_in, const int* in_sizes, int n_in,
                              void* d_out, int out_size, void* d_ws, size_t ws_size,
                              hipStream_t stream) {
  (void)in_sizes; (void)n_in; (void)out_size; (void)ws_size;
  const int*   x     = (const int*)  d_in[0];
  const float* embed = (const float*)d_in[1];
  const float* Wg    = (const float*)d_in[2];
  const float* bg    = (const float*)d_in[3];
  const float* W1    = (const float*)d_in[4];
  const float* b1    = (const float*)d_in[5];
  const float* W2    = (const float*)d_in[6];
  const float* b2    = (const float*)d_in[7];
  const float* gamma = (const float*)d_in[8];
  const float* beta  = (const float*)d_in[9];
  const float* Wh    = (const float*)d_in[10];
  const float* bh    = (const float*)d_in[11];
  float* out = (float*)d_out;

  char* ws = (char*)d_ws;
  size_t off = 0;
  auto carve = [&](size_t bytes) -> void* {
    void* p = ws + off;
    off = (off + bytes + 255) & ~(size_t)255;
    return p;
  };
  bf16*  w1t      = (bf16*) carve((size_t)NEXP * HIDDEN * DMODEL * 2);   // 33.6 MB
  bf16*  w2t      = (bf16*) carve((size_t)NEXP * DMODEL * HIDDEN * 2);   // 33.6 MB
  bf16*  whT      = (bf16*) carve((size_t)VOCAB * DMODEL * 2);           // 65.5 MB
  bf16*  hbf      = (bf16*) carve((size_t)NTOK * DMODEL * 2);            // 4 MB
  bf16*  lnbf     = (bf16*) carve((size_t)NTOK * DMODEL * 2);            // 4 MB
  float* combined = (float*)carve((size_t)NTOK * DMODEL * 4);            // 8 MB
  float* gatew    = (float*)carve((size_t)NTOK * NEXP * 4);              // 64 KB

  const dim3 blk256(256), blk16x16(16, 16);

  // weight transpose-converts (independent; L2-resident thereafter)
  moe_transpose_cvt<<<dim3(HIDDEN / 16, DMODEL / 16, NEXP), blk16x16, 0, stream>>>(
      W1, w1t, DMODEL, HIDDEN);
  moe_transpose_cvt<<<dim3(DMODEL / 16, HIDDEN / 16, NEXP), blk16x16, 0, stream>>>(
      W2, w2t, HIDDEN, DMODEL);
  moe_transpose_cvt<<<dim3(VOCAB / 16, DMODEL / 16, 1), blk16x16, 0, stream>>>(
      Wh, whT, DMODEL, VOCAB);

  moe_embed_gate<<<dim3(NTOK), blk256, 0, stream>>>(x, embed, Wg, bg, hbf, gatew);

  hipMemsetAsync(combined, 0, (size_t)NTOK * DMODEL * 4, stream);

  moe_ffn<<<dim3(NTOK / 16, NEXP), blk256, 0, stream>>>(
      hbf, w1t, w2t, b1, b2, gatew, combined);

  moe_layernorm<<<dim3(NTOK), blk256, 0, stream>>>(combined, gamma, beta, lnbf);

  moe_head<<<dim3(NTOK / 128, VOCAB / 256), blk256, 0, stream>>>(lnbf, whT, bh, out);
}
